// SupNet_30296699306320
// MI455X (gfx1250) — matrix operand order repaired and checked by reading, not compiled, not run on hardware
//
#include <hip/hip_runtime.h>
#include <hip/hip_bf16.h>

// ---------- types ----------
typedef __attribute__((ext_vector_type(16))) _Float16 v16h;
typedef __attribute__((ext_vector_type(4)))  _Float16 v4h;
typedef __attribute__((ext_vector_type(8)))  float    v8f;
typedef __attribute__((ext_vector_type(4)))  float    f32x4;
typedef __attribute__((ext_vector_type(4)))  unsigned u32x4;
typedef __attribute__((ext_vector_type(4)))  int      v4i;
typedef __attribute__((address_space(1))) v4i* g_v4i_p;   // global AS pointer to int4
typedef __attribute__((address_space(3))) v4i* l_v4i_p;   // LDS AS pointer to int4

#define NCELL 768
#define DEMB  128
#define C1    256   // hidden channels layer 1
#define C2    128   // hidden channels layer 2
#define BJ    128   // j-tile (pairs) per workgroup
#define HSTRIDE 272 // 256 + 16 halves pad (bank stagger), 544 B row stride

// LDS layout (bytes)
#define OFF_W2   0                        // 128*256 half = 65536
#define OFF_H    65536                    // 128*272 half = 69632
#define OFF_CPB  (65536 + 69632)          // 256 f32      = 1024
#define OFF_OUT  (65536 + 69632 + 1024)   // 128*2 f32    = 1024
#define DYN_LDS  (65536 + 69632 + 1024 + 1024)

// ---------- kernel 1: RP[j,o] = z[j,:]·W1[o,:128], CP[i,o] = z[i,:]·W1[o,128:] ----------
__global__ __launch_bounds__(256) void precompute_rp_cp(
    const float* __restrict__ z, const float* __restrict__ W1,
    float* __restrict__ RP, float* __restrict__ CP) {
  int gid  = blockIdx.x * 256 + threadIdx.x;      // 2 * 768 * 256 total
  int half = gid / (NCELL * C1);
  int rem  = gid - half * (NCELL * C1);
  int row  = rem >> 8;
  int o    = rem & 255;
  const f32x4* zr = (const f32x4*)(z  + row * DEMB);
  const f32x4* wr = (const f32x4*)(W1 + o * (2 * DEMB) + half * DEMB);
  float s = 0.f;
#pragma unroll 8
  for (int t = 0; t < DEMB / 4; ++t) {
    f32x4 a = zr[t], b = wr[t];
    s += a.x * b.x + a.y * b.y + a.z * b.z + a.w * b.w;
  }
  (half ? CP : RP)[row * C1 + o] = s;
}

// ---------- kernel 1b: W2 f32 -> f16 ----------
__global__ __launch_bounds__(256) void convert_w2(
    const float* __restrict__ W2, _Float16* __restrict__ W2h) {
  int gid = blockIdx.x * 256 + threadIdx.x;       // 128*256 total
  W2h[gid] = (_Float16)W2[gid];
}

// ---------- kernel 2: fused pair GEMM + final layer ----------
__global__ __launch_bounds__(256) void pairdist_kernel(
    const float* __restrict__ RP, const float* __restrict__ CP,
    const _Float16* __restrict__ W2h,
    const float* __restrict__ b1, const float* __restrict__ b2,
    const float* __restrict__ Wf, const float* __restrict__ bf,
    float* __restrict__ out) {
  extern __shared__ char smem[];
  _Float16* sW2 = (_Float16*)(smem + OFF_W2);   // [128][256] half, B operand source
  _Float16* sH  = (_Float16*)(smem + OFF_H);    // [128][HSTRIDE] half, A operand source
  float*   sCPB = (float*)(smem + OFF_CPB);     // col_part[i] + b1
  float*   sOut = (float*)(smem + OFF_OUT);     // [128][2] final accum

  const int tid = threadIdx.x;
  const int i   = blockIdx.y;        // column-cell index (fixed per WG)
  const int j0  = blockIdx.x * BJ;   // pair tile base

  // --- stage W2 (f16) into LDS: 4096 x 16B chunks via async-to-LDS ---
#if __has_builtin(__builtin_amdgcn_global_load_async_to_lds_b128)
  {
    const _Float16* srcp = W2h + tid * 8;   // 16 B per lane per issue
    _Float16*       dstp = sW2 + tid * 8;
#pragma unroll
    for (int t = 0; t < 16; ++t) {
      __builtin_amdgcn_global_load_async_to_lds_b128(
          (g_v4i_p)(uintptr_t)(srcp + 2048 * t),
          (l_v4i_p)(unsigned)(uintptr_t)(dstp + 2048 * t), 0, 0);
    }
    asm volatile("s_wait_asynccnt 0x0" ::: "memory");
  }
#else
  {
    const u32x4* src = (const u32x4*)W2h;
    u32x4*       dst = (u32x4*)sW2;
#pragma unroll
    for (int t = 0; t < 16; ++t) dst[tid + 256 * t] = src[tid + 256 * t];
  }
#endif
  // --- cpb[o] = CP[i][o] + b1[o] ---
  sCPB[tid] = CP[i * C1 + tid] + b1[tid];
  sOut[tid] = bf[tid & 1];
  __syncthreads();

  // --- build h tile in f16: h[r][c] = relu(RP[j0+r][c] + cpb[c]) ---
  {
    int r  = tid >> 1;
    int c0 = (tid & 1) * 128;
    const float* rp   = RP + (j0 + r) * C1 + c0;
    _Float16*    hrow = sH + r * HSTRIDE + c0;
#pragma unroll 4
    for (int c = 0; c < 128; c += 4) {
      f32x4 x  = *(const f32x4*)(rp + c);
      f32x4 cb = *(const f32x4*)(sCPB + c0 + c);
      v4h y;
      y.x = (_Float16)fmaxf(x.x + cb.x, 0.f);
      y.y = (_Float16)fmaxf(x.y + cb.y, 0.f);
      y.z = (_Float16)fmaxf(x.z + cb.z, 0.f);
      y.w = (_Float16)fmaxf(x.w + cb.w, 0.f);
      *(v4h*)(hrow + c) = y;
    }
  }
  __syncthreads();

  // --- wave tiling: 4 waves over M (2x16 pairs each -> 128), 2 waves over N (64 ch each) ---
  const int lane = tid & 31;
  const int wave = tid >> 5;
  const int mgrp = wave & 3;     // rows [mgrp*32, mgrp*32+32)
  const int nsub = wave >> 2;    // channels [nsub*64, nsub*64+64)
  const int lhi  = lane >> 4;
  const int llo  = lane & 15;

  // A fragment source rows (16-bit A 16x32 layout: lane half splits K ranges)
  const _Float16* aRow0 = sH + (mgrp * 32 + llo)      * HSTRIDE + lhi * 8;
  const _Float16* aRow1 = sH + (mgrp * 32 + 16 + llo) * HSTRIDE + lhi * 8;

  union Frag { v16h v; u32x4 q[2]; };
  v8f acc[2][4] = {};

#pragma unroll
  for (int kb = 0; kb < 8; ++kb) {
    const int k0 = kb * 32;
    Frag a0, a1;
    a0.q[0] = *(const u32x4*)(aRow0 + k0);
    a0.q[1] = *(const u32x4*)(aRow0 + k0 + 16);
    a1.q[0] = *(const u32x4*)(aRow1 + k0);
    a1.q[1] = *(const u32x4*)(aRow1 + k0 + 16);
#pragma unroll
    for (int nt = 0; nt < 4; ++nt) {
      const int n = nsub * 64 + nt * 16 + llo;  // B lane = output channel
      const _Float16* p = sW2 + n * C1 + k0 + lhi * 8;
      Frag b;
      b.q[0] = *(const u32x4*)(p);
      b.q[1] = *(const u32x4*)(p + 16);
      acc[0][nt] = __builtin_amdgcn_wmma_f32_16x16x32_f16(
          false, a0.v, false, b.v, (short)0, acc[0][nt], false, false);
      acc[1][nt] = __builtin_amdgcn_wmma_f32_16x16x32_f16(
          false, a1.v, false, b.v, (short)0, acc[1][nt], false, false);
    }
  }

  // --- epilogue: h2 = relu(acc + b2); dists partial = h2 · Wf^T ---
  float wf0[4], wf1[4], bb[4];
#pragma unroll
  for (int nt = 0; nt < 4; ++nt) {
    int n = nsub * 64 + nt * 16 + llo;
    wf0[nt] = Wf[n];
    wf1[nt] = Wf[C2 + n];
    bb[nt]  = b2[n];
  }
  float s0[16], s1[16];
#pragma unroll
  for (int mt = 0; mt < 2; ++mt) {
#pragma unroll
    for (int v = 0; v < 8; ++v) {
      float t0 = 0.f, t1 = 0.f;
#pragma unroll
      for (int nt = 0; nt < 4; ++nt) {
        float x = fmaxf(acc[mt][nt][v] + bb[nt], 0.f);
        t0 += x * wf0[nt];
        t1 += x * wf1[nt];
      }
      s0[mt * 8 + v] = t0;
      s1[mt * 8 + v] = t1;
    }
  }
  // reduce across the 16 lanes of each lane-half (rows differ per half)
#pragma unroll
  for (int m = 1; m <= 8; m <<= 1) {
#pragma unroll
    for (int v = 0; v < 16; ++v) {
      s0[v] += __shfl_xor(s0[v], m, 32);
      s1[v] += __shfl_xor(s1[v], m, 32);
    }
  }
  if (llo == 0) {
#pragma unroll
    for (int mt = 0; mt < 2; ++mt) {
#pragma unroll
      for (int v = 0; v < 8; ++v) {
        int row = mgrp * 32 + mt * 16 + v + lhi * 8;  // C/D layout: M = v + 8*lane_half
        atomicAdd(&sOut[row * 2 + 0], s0[mt * 8 + v]);
        atomicAdd(&sOut[row * 2 + 1], s1[mt * 8 + v]);
      }
    }
  }
  __syncthreads();
  out[(i * NCELL + j0) * 2 + tid] = sOut[tid];   // 256 contiguous f32 per WG
}

// ---------- host ----------
extern "C" void kernel_launch(void* const* d_in, const int* in_sizes, int n_in,
                              void* d_out, int out_size, void* d_ws, size_t ws_size,
                              hipStream_t stream) {
  const float* z  = (const float*)d_in[0];
  const float* W1 = (const float*)d_in[1];
  const float* b1 = (const float*)d_in[2];
  const float* W2 = (const float*)d_in[3];
  const float* b2 = (const float*)d_in[4];
  const float* Wf = (const float*)d_in[5];
  const float* bf = (const float*)d_in[6];
  float* out = (float*)d_out;

  char* ws = (char*)d_ws;
  float*    RP  = (float*)(ws);                          // 768*256 f32
  float*    CP  = (float*)(ws + 786432);                 // 768*256 f32
  _Float16* W2h = (_Float16*)(ws + 1572864);             // 128*256 f16

  (void)hipFuncSetAttribute((const void*)pairdist_kernel,
                            hipFuncAttributeMaxDynamicSharedMemorySize, DYN_LDS);

  precompute_rp_cp<<<(2 * NCELL * C1) / 256, 256, 0, stream>>>(z, W1, RP, CP);
  convert_w2<<<(C2 * C1) / 256, 256, 0, stream>>>(W2, W2h);

  dim3 grid(NCELL / BJ, NCELL);   // 6 x 768 workgroups
  pairdist_kernel<<<grid, 256, DYN_LDS, stream>>>(RP, CP, W2h, b1, b2, Wf, bf, out);
}
